// TransformerDecoder_25666724561143
// MI455X (gfx1250) — matrix operand order repaired and checked by reading, not compiled, not run on hardware
//
#include <hip/hip_runtime.h>
#include <hip/hip_bf16.h>

typedef __bf16 v8bf  __attribute__((ext_vector_type(8)));
typedef __bf16 v16bf __attribute__((ext_vector_type(16)));
typedef float  v8f   __attribute__((ext_vector_type(8)));
typedef int    v4i   __attribute__((vector_size(16)));

#if defined(__HIP_DEVICE_COMPILE__) && defined(__has_builtin)
#  if __has_builtin(__builtin_amdgcn_global_load_async_to_lds_b128) && \
      __has_builtin(__builtin_amdgcn_s_wait_asynccnt)
#    define USE_ASYNC_LDS 1
#  endif
#endif
#ifndef USE_ASYNC_LDS
#  define USE_ASYNC_LDS 0
#endif

#define BATCH  32
#define NSEQ   144
#define EIN    127
#define EMBD   512
#define NH     8
#define DKH    64
#define FDIM   16384
#define NLAYER 4
#define MROWS  (BATCH * NSEQ)   /* 4608 */
#define FCHUNK 2048
#define LNEPS  1e-5f
#define AKP    72               /* padded LDS row: 64 bf16 -> 72 (144B, 16B-aligned) */

#define GF_BIAS    1
#define GF_RELU    2
#define GF_ACC     4
#define GF_F32OUT  8
#define GF_BF16OUT 16

typedef __attribute__((address_space(1))) v4i* g4_ptr;   // global int4*
typedef __attribute__((address_space(3))) v4i* l4_ptr;   // LDS int4*

// ---------------------------------------------------------------------------
// GEMM: C[M,N] = A[M,K] @ B[K,N] with B given TRANSPOSED as BT[n][k] (bf16).
// A bf16, accumulate fp32 via V_WMMA_F32_16X16X32_BF16.
// 256 threads = 8 waves; block tile 128x128, K-step 64, double-buffered LDS.
// Preferred staging: GLOBAL_LOAD_ASYNC_TO_LDS_B128 (no VGPR transit, ASYNCcnt
// overlap with the WMMA block). Fallback / EDGE: compute-first then
// load+store with minimal register live range (no scratch spills).
// ---------------------------------------------------------------------------
template <bool EDGE>
__global__ __launch_bounds__(256) __attribute__((amdgpu_waves_per_eu(1, 8)))
void gemm_wmma_bf16(
    const __bf16* __restrict__ A, int lda,
    const __bf16* __restrict__ BT, int ldb,
    const float* __restrict__ bias,
    float* __restrict__ C, __bf16* __restrict__ Cb, int ldc,
    int Md, int Nd, int Kd, int flags)
{
  (void)Md;
  __shared__ __bf16 As[2][128][AKP];
  __shared__ __bf16 Bt[2][128][AKP];

  const int m0   = blockIdx.y * 128;
  const int n0   = blockIdx.x * 128;
  const int t    = threadIdx.x;
  const int lane = t & 31;
  const int wave = t >> 5;
  const int wm   = wave >> 1;     // 0..3 : M position of wave
  const int wn   = wave & 1;      // 0..1 : N position of wave
  const int half = lane >> 4;     // 0|1  : lane half (K-run select)
  const int l16  = lane & 15;

  // staging: tile is 128 rows x 64 K; thread covers one row-half (32 K = 4 x b128)
  const int sr = t >> 1;          // 0..127
  const int sk = (t & 1) * 32;

  v8f acc[2][4] = {};

  auto computeTile = [&](int buf) {
    #pragma unroll
    for (int ks = 0; ks < 2; ++ks) {   // two K=32 WMMA steps per LDS tile
      v16bf af[2];
      // A 16x32 frag: lanes 0-15: K 0-7 & 16-23 ; lanes 16-31: K 8-15 & 24-31
      #pragma unroll
      for (int ti = 0; ti < 2; ++ti) {
        int row = wm * 32 + ti * 16 + l16;
        v8bf lo = *(const v8bf*)&As[buf][row][ks * 32 + half * 8];
        v8bf hi = *(const v8bf*)&As[buf][row][ks * 32 + 16 + half * 8];
        af[ti] = __builtin_shufflevector(lo, hi, 0,1,2,3,4,5,6,7,8,9,10,11,12,13,14,15);
      }
      // B 32x16 frag: lanes 0-15: K 0-7 & 16-23 ; lanes 16-31: K 8-15 & 24-31, N = lane%16
      #pragma unroll
      for (int tj = 0; tj < 4; ++tj) {
        int col = wn * 64 + tj * 16 + l16;
        v8bf lo = *(const v8bf*)&Bt[buf][col][ks * 32 + half * 8];
        v8bf hi = *(const v8bf*)&Bt[buf][col][ks * 32 + 16 + half * 8];
        v16bf bfr = __builtin_shufflevector(lo, hi, 0,1,2,3,4,5,6,7,8,9,10,11,12,13,14,15);
        acc[0][tj] = __builtin_amdgcn_wmma_f32_16x16x32_bf16(
            false, af[0], false, bfr, (short)0, acc[0][tj], false, false);
        acc[1][tj] = __builtin_amdgcn_wmma_f32_16x16x32_bf16(
            false, af[1], false, bfr, (short)0, acc[1][tj], false, false);
      }
    }
  };

  const int nk = Kd >> 6;

#if USE_ASYNC_LDS
  if (!EDGE) {
    // ---- async global->LDS staging: no VGPR transit, ASYNCcnt overlap ----
    auto asyncTile = [&](int k0, int nb) {
      #pragma unroll
      for (int j = 0; j < 4; ++j) {
        __bf16* src = const_cast<__bf16*>(&A[(size_t)(m0 + sr) * lda + k0 + sk + j * 8]);
        __builtin_amdgcn_global_load_async_to_lds_b128(
            (g4_ptr)(void*)src, (l4_ptr)(void*)&As[nb][sr][sk + j * 8], 0, 0);
      }
      #pragma unroll
      for (int j = 0; j < 4; ++j) {
        __bf16* src = const_cast<__bf16*>(&BT[(size_t)(n0 + sr) * ldb + k0 + sk + j * 8]);
        __builtin_amdgcn_global_load_async_to_lds_b128(
            (g4_ptr)(void*)src, (l4_ptr)(void*)&Bt[nb][sr][sk + j * 8], 0, 0);
      }
    };
    asyncTile(0, 0);
    __builtin_amdgcn_s_wait_asynccnt(0);
    __syncthreads();
    for (int kb = 0; kb < nk; ++kb) {
      const int buf   = kb & 1;
      const bool more = (kb + 1 < nk);
      if (more) asyncTile((kb + 1) << 6, buf ^ 1);  // DMA next tile during WMMAs
      computeTile(buf);
      if (more) __builtin_amdgcn_s_wait_asynccnt(0);
      __syncthreads();
    }
  } else
#endif
  {
    // ---- fallback: stage via registers with minimal live range ----
    auto stageTile = [&](int k0, int nb) {
      uint4 aReg[4], bReg[4];
      const uint4 zero4 = {0u, 0u, 0u, 0u};
      #pragma unroll
      for (int j = 0; j < 4; ++j)
        aReg[j] = *(const uint4*)&A[(size_t)(m0 + sr) * lda + k0 + sk + j * 8];
      if (EDGE) {
        bool ok = (n0 + sr) < Nd;
        #pragma unroll
        for (int j = 0; j < 4; ++j)
          bReg[j] = ok ? *(const uint4*)&BT[(size_t)(n0 + sr) * ldb + k0 + sk + j * 8]
                       : zero4;
      } else {
        #pragma unroll
        for (int j = 0; j < 4; ++j)
          bReg[j] = *(const uint4*)&BT[(size_t)(n0 + sr) * ldb + k0 + sk + j * 8];
      }
      #pragma unroll
      for (int j = 0; j < 4; ++j) *(uint4*)&As[nb][sr][sk + j * 8] = aReg[j];
      #pragma unroll
      for (int j = 0; j < 4; ++j) *(uint4*)&Bt[nb][sr][sk + j * 8] = bReg[j];
    };
    stageTile(0, 0);
    __syncthreads();
    for (int kb = 0; kb < nk; ++kb) {
      const int buf   = kb & 1;
      const bool more = (kb + 1 < nk);
      computeTile(buf);
      if (more) stageTile((kb + 1) << 6, buf ^ 1);  // short live range: no spill
      __syncthreads();
    }
  }

  // ---- epilogue: C/D layout => VGPR p: lanes 0-15 M=p, lanes 16-31 M=p+8 ----
  #pragma unroll
  for (int ti = 0; ti < 2; ++ti) {
    #pragma unroll
    for (int tj = 0; tj < 4; ++tj) {
      int col = n0 + wn * 64 + tj * 16 + l16;
      if (EDGE && col >= Nd) continue;
      int rbase = m0 + wm * 32 + ti * 16 + half * 8;
      #pragma unroll
      for (int p = 0; p < 8; ++p) {
        size_t idx = (size_t)(rbase + p) * ldc + col;
        float v = acc[ti][tj][p];
        if (flags & GF_ACC)  v += C[idx];
        if (flags & GF_BIAS) v += bias[col];
        if (flags & GF_RELU) v = fmaxf(v, 0.0f);
        if (flags & GF_F32OUT)  C[idx]  = v;
        if (flags & GF_BF16OUT) Cb[idx] = (__bf16)v;
      }
    }
  }
}

// ---------------------------------------------------------------------------
// Transpose + cast: W fp32 [K][N] -> WT bf16 [N][K], LDS-tiled 64x64.
// ---------------------------------------------------------------------------
__global__ __launch_bounds__(256) void transpose_cast_kernel(
    const float* __restrict__ W, __bf16* __restrict__ WT, int K, int N)
{
  __shared__ float tile[64][65];
  const int kt = blockIdx.y * 64;
  const int nt = blockIdx.x * 64;
  const int t  = threadIdx.x;
  const int r  = t >> 6;   // 0..3
  const int c  = t & 63;
  #pragma unroll
  for (int j = 0; j < 16; ++j) {
    int kk = kt + r + j * 4;
    int nn = nt + c;
    tile[r + j * 4][c] = (kk < K && nn < N) ? W[(size_t)kk * N + nn] : 0.0f;
  }
  __syncthreads();
  #pragma unroll
  for (int j = 0; j < 16; ++j) {
    int nn = nt + r + j * 4;
    int kk = kt + c;
    if (nn < N && kk < K)
      WT[(size_t)nn * K + kk] = (__bf16)tile[c][r + j * 4];
  }
}

// ---------------------------------------------------------------------------
// Build XP = concat(x, positional channel) as bf16  [4608 x 128]
// ---------------------------------------------------------------------------
__global__ __launch_bounds__(256) void build_xp_kernel(
    const float* __restrict__ x, __bf16* __restrict__ xp)
{
  int i = blockIdx.x * 256 + threadIdx.x;
  if (i >= MROWS * 128) return;
  int row = i >> 7, c = i & 127;
  float v;
  if (c < EIN) {
    v = x[(size_t)row * EIN + c];
  } else {
    int n = row % NSEQ;
    v = (float)n / (float)(NSEQ - 1);
  }
  xp[i] = (__bf16)v;
}

// ---------------------------------------------------------------------------
// Diagonal-softmax attention (fp32 in, bf16 out):
// out[b,n,h,:] = softmax_m(q_n . k_m)[n] * v[b,n,h,:]   (optional causal mask)
// ---------------------------------------------------------------------------
__global__ __launch_bounds__(256) void attn_diag_kernel(
    const float* __restrict__ Q, const float* __restrict__ K,
    const float* __restrict__ V, __bf16* __restrict__ Out, int masked)
{
  __shared__ float kb[NSEQ][DKH + 1];

  const int bh = blockIdx.x;
  const int b  = bh / NH;
  const int h  = bh % NH;
  const size_t base = (size_t)(b * NSEQ) * EMBD + h * DKH;

  for (int i = threadIdx.x; i < NSEQ * DKH; i += 256) {
    int r = i >> 6, c = i & 63;
    kb[r][c] = K[base + (size_t)r * EMBD + c];
  }
  __syncthreads();

  const int lane = threadIdx.x & 31;
  const int wave = threadIdx.x >> 5;

  for (int n = wave; n < NSEQ; n += 8) {
    const int mlim = masked ? (n + 1) : NSEQ;
    float q0 = Q[base + (size_t)n * EMBD + lane];
    float q1 = Q[base + (size_t)n * EMBD + 32 + lane];

    float s[5];
    float smax = -3.0e38f;
    #pragma unroll
    for (int i = 0; i < 5; ++i) {
      int m = lane + (i << 5);
      bool valid = (m < mlim);
      int mm = valid ? m : 0;
      float acc = 0.0f;
      #pragma unroll
      for (int k2 = 0; k2 < DKH; ++k2) {
        float qv = __shfl((k2 < 32) ? q0 : q1, k2 & 31, 32);
        acc += qv * kb[mm][k2];
      }
      s[i] = valid ? acc : -3.0e38f;
      smax = fmaxf(smax, s[i]);
    }
    #pragma unroll
    for (int off = 16; off; off >>= 1)
      smax = fmaxf(smax, __shfl_xor(smax, off, 32));

    float sum = 0.0f, diag = 0.0f;
    #pragma unroll
    for (int i = 0; i < 5; ++i) {
      int m = lane + (i << 5);
      bool valid = (m < mlim);
      float e = valid ? __expf(s[i] - smax) : 0.0f;
      sum += e;
      if (m == n) diag = e;
    }
    #pragma unroll
    for (int off = 16; off; off >>= 1) {
      sum  += __shfl_xor(sum, off, 32);
      diag += __shfl_xor(diag, off, 32);
    }
    float dval = diag / sum;

    #pragma unroll
    for (int c = 0; c < DKH; c += 32)
      Out[base + (size_t)n * EMBD + c + lane] =
          (__bf16)(dval * V[base + (size_t)n * EMBD + c + lane]);
  }
}

// ---------------------------------------------------------------------------
// y = LayerNorm(a + r) * g + b  — one wave per 512-wide row; fp32 + bf16 out
// ---------------------------------------------------------------------------
__global__ __launch_bounds__(256) void resln_kernel(
    const float* __restrict__ a, const float* __restrict__ r,
    const float* __restrict__ g, const float* __restrict__ bb,
    float* __restrict__ y, __bf16* __restrict__ yb)
{
  const int row  = blockIdx.x * 8 + (threadIdx.x >> 5);
  const int lane = threadIdx.x & 31;
  const size_t off = (size_t)row * EMBD;

  float vals[16];
  float sum = 0.0f;
  #pragma unroll
  for (int i = 0; i < 16; ++i) {
    int c = lane + i * 32;
    float v = a[off + c] + r[off + c];
    vals[i] = v;
    sum += v;
  }
  #pragma unroll
  for (int o = 16; o; o >>= 1) sum += __shfl_xor(sum, o, 32);
  float mu = sum * (1.0f / EMBD);

  float vs = 0.0f;
  #pragma unroll
  for (int i = 0; i < 16; ++i) {
    float d = vals[i] - mu;
    vs += d * d;
  }
  #pragma unroll
  for (int o = 16; o; o >>= 1) vs += __shfl_xor(vs, o, 32);
  float rs = rsqrtf(vs * (1.0f / EMBD) + LNEPS);

  #pragma unroll
  for (int i = 0; i < 16; ++i) {
    int c = lane + i * 32;
    float ov = (vals[i] - mu) * rs * g[c] + bb[c];
    y[off + c]  = ov;
    yb[off + c] = (__bf16)ov;
  }
}

// ---------------------------------------------------------------------------
extern "C" void kernel_launch(void* const* d_in, const int* in_sizes, int n_in,
                              void* d_out, int out_size, void* d_ws, size_t ws_size,
                              hipStream_t stream)
{
  (void)in_sizes; (void)n_in; (void)out_size; (void)ws_size;

  const float* x    = (const float*)d_in[1];
  const float* Wenc = (const float*)d_in[3];
  const float* benc = (const float*)d_in[4];
  const float* Wq   = (const float*)d_in[5];
  const float* Wk   = (const float*)d_in[6];
  const float* Wv   = (const float*)d_in[7];
  const float* Wo   = (const float*)d_in[8];
  const float* bo   = (const float*)d_in[9];
  const float* ln1g = (const float*)d_in[10];
  const float* ln1b = (const float*)d_in[11];
  const float* ln2g = (const float*)d_in[12];
  const float* ln2b = (const float*)d_in[13];
  const float* ln3g = (const float*)d_in[14];
  const float* ln3b = (const float*)d_in[15];
  const float* W1   = (const float*)d_in[16];
  const float* b1   = (const float*)d_in[17];
  const float* W2   = (const float*)d_in[18];
  const float* b2   = (const float*)d_in[19];
  const float* Wdec = (const float*)d_in[20];
  const float* bdec = (const float*)d_in[21];

  char* base = (char*)d_ws;
  size_t off = 0;
  auto allocf = [&](size_t n) { float* p = (float*)(base + off);  off += n * 4; off = (off + 255) & ~(size_t)255; return p; };
  auto allocb = [&](size_t n) { __bf16* p = (__bf16*)(base + off); off += n * 2; off = (off + 255) & ~(size_t)255; return p; };

  const size_t SZ = (size_t)MROWS * EMBD;
  // fp32 activations
  float* XE  = allocf(SZ);
  float* OUT = allocf(SZ);
  float* QLN = allocf(SZ);
  float* ATT = allocf(SZ);
  float* AO  = allocf(SZ);
  float* FB  = allocf(SZ);
  float* Qf  = allocf(SZ);
  float* Kf  = allocf(SZ);
  float* Vf  = allocf(SZ);
  // bf16 activations
  __bf16* XPb  = allocb((size_t)MROWS * 128);
  __bf16* XEb  = allocb(SZ);
  __bf16* OUTb = allocb(SZ);
  __bf16* QLNb = allocb(SZ);
  __bf16* ATTb = allocb(SZ);
  __bf16* AOb  = allocb(SZ);
  __bf16* DVb  = allocb(SZ);
  __bf16* Hcb  = allocb((size_t)MROWS * FCHUNK);
  // bf16 transposed weights
  __bf16* WencT = allocb((size_t)(EIN + 1) * EMBD);
  __bf16* WqT   = allocb((size_t)NLAYER * EMBD * EMBD);
  __bf16* WkT   = allocb((size_t)NLAYER * EMBD * EMBD);
  __bf16* WvT   = allocb((size_t)NLAYER * EMBD * EMBD);
  __bf16* WoT   = allocb((size_t)NLAYER * EMBD * EMBD);
  __bf16* W1T   = allocb((size_t)NLAYER * EMBD * FDIM);
  __bf16* W2T   = allocb((size_t)NLAYER * FDIM * EMBD);
  __bf16* WdecT = allocb((size_t)EIN * EMBD);

  auto tcast = [&](const float* W, __bf16* WT, int K, int N) {
    dim3 grid((N + 63) / 64, (K + 63) / 64);
    transpose_cast_kernel<<<grid, 256, 0, stream>>>(W, WT, K, N);
  };
  auto gemm = [&](const __bf16* A, int lda, const __bf16* BT, int ldb,
                  const float* bias, float* C, __bf16* Cb, int ldc,
                  int Md, int Nd, int Kd, int flags) {
    dim3 grid((Nd + 127) / 128, Md / 128);
    if ((Nd & 127) == 0)
      gemm_wmma_bf16<false><<<grid, 256, 0, stream>>>(A, lda, BT, ldb, bias, C, Cb, ldc, Md, Nd, Kd, flags);
    else
      gemm_wmma_bf16<true><<<grid, 256, 0, stream>>>(A, lda, BT, ldb, bias, C, Cb, ldc, Md, Nd, Kd, flags);
  };

  // ----- weight transpose+cast (per call; stateless) -----
  tcast(Wenc, WencT, EIN + 1, EMBD);
  tcast(Wdec, WdecT, EMBD, EIN);
  for (int l = 0; l < NLAYER; ++l) {
    tcast(Wq + (size_t)l * EMBD * EMBD, WqT + (size_t)l * EMBD * EMBD, EMBD, EMBD);
    tcast(Wk + (size_t)l * EMBD * EMBD, WkT + (size_t)l * EMBD * EMBD, EMBD, EMBD);
    tcast(Wv + (size_t)l * EMBD * EMBD, WvT + (size_t)l * EMBD * EMBD, EMBD, EMBD);
    tcast(Wo + (size_t)l * EMBD * EMBD, WoT + (size_t)l * EMBD * EMBD, EMBD, EMBD);
    tcast(W1 + (size_t)l * EMBD * FDIM, W1T + (size_t)l * EMBD * FDIM, EMBD, FDIM);
    tcast(W2 + (size_t)l * FDIM * EMBD, W2T + (size_t)l * FDIM * EMBD, FDIM, EMBD);
  }

  // ----- positional channel + encoder projection -----
  build_xp_kernel<<<(MROWS * 128) / 256, 256, 0, stream>>>(x, XPb);
  gemm(XPb, 128, WencT, 128, benc, XE, XEb, EMBD,
       MROWS, EMBD, 128, GF_BIAS | GF_F32OUT | GF_BF16OUT);

  const float*  cur  = XE;
  const __bf16* curb = XEb;
  for (int l = 0; l < NLAYER; ++l) {
    const __bf16* WqTl = WqT + (size_t)l * EMBD * EMBD;
    const __bf16* WkTl = WkT + (size_t)l * EMBD * EMBD;
    const __bf16* WvTl = WvT + (size_t)l * EMBD * EMBD;
    const __bf16* WoTl = WoT + (size_t)l * EMBD * EMBD;
    const __bf16* W1Tl = W1T + (size_t)l * EMBD * FDIM;
    const __bf16* W2Tl = W2T + (size_t)l * FDIM * EMBD;
    const float* bol = bo + (size_t)l * EMBD;
    const float* b1l = b1 + (size_t)l * FDIM;
    const float* b2l = b2 + (size_t)l * EMBD;

    // ----- masked self-attention on `cur` -----
    gemm(curb, EMBD, WqTl, EMBD, nullptr, Qf, nullptr, EMBD, MROWS, EMBD, EMBD, GF_F32OUT);
    gemm(curb, EMBD, WkTl, EMBD, nullptr, Kf, nullptr, EMBD, MROWS, EMBD, EMBD, GF_F32OUT);
    gemm(curb, EMBD, WvTl, EMBD, nullptr, Vf, nullptr, EMBD, MROWS, EMBD, EMBD, GF_F32OUT);
    attn_diag_kernel<<<BATCH * NH, 256, 0, stream>>>(Qf, Kf, Vf, DVb, 1);
    gemm(DVb, EMBD, WoTl, EMBD, bol, ATT, nullptr, EMBD,
         MROWS, EMBD, EMBD, GF_BIAS | GF_F32OUT);
    // q = LN(a1 + xe)
    resln_kernel<<<MROWS / 8, 256, 0, stream>>>(ATT, XE, ln1g + l * EMBD, ln1b + l * EMBD, QLN, QLNb);

    // ----- second attention: Q from q, K/V from xe, no mask -----
    gemm(QLNb, EMBD, WqTl, EMBD, nullptr, Qf, nullptr, EMBD, MROWS, EMBD, EMBD, GF_F32OUT);
    gemm(XEb,  EMBD, WkTl, EMBD, nullptr, Kf, nullptr, EMBD, MROWS, EMBD, EMBD, GF_F32OUT);
    gemm(XEb,  EMBD, WvTl, EMBD, nullptr, Vf, nullptr, EMBD, MROWS, EMBD, EMBD, GF_F32OUT);
    attn_diag_kernel<<<BATCH * NH, 256, 0, stream>>>(Qf, Kf, Vf, DVb, 0);
    gemm(DVb, EMBD, WoTl, EMBD, bol, ATT, ATTb, EMBD,
         MROWS, EMBD, EMBD, GF_BIAS | GF_F32OUT | GF_BF16OUT);  // a2

    // ----- ff(a2) -> FB, split-K over F (bf16 H, fp32 accumulate in FB) -----
    for (int fc = 0; fc < FDIM; fc += FCHUNK) {
      gemm(ATTb, EMBD, W1Tl + (size_t)fc * EMBD, EMBD, b1l + fc, nullptr, Hcb, FCHUNK,
           MROWS, FCHUNK, EMBD, GF_BIAS | GF_RELU | GF_BF16OUT);
      gemm(Hcb, FCHUNK, W2Tl + fc, FDIM, b2l, FB, nullptr, EMBD,
           MROWS, EMBD, FCHUNK, GF_F32OUT | ((fc == 0) ? GF_BIAS : GF_ACC));
    }
    // ao = LN(ff + a2)
    resln_kernel<<<MROWS / 8, 256, 0, stream>>>(FB, ATT, ln2g + l * EMBD, ln2b + l * EMBD, AO, AOb);

    // ----- ff2(ao) -> FB -----
    for (int fc = 0; fc < FDIM; fc += FCHUNK) {
      gemm(AOb, EMBD, W1Tl + (size_t)fc * EMBD, EMBD, b1l + fc, nullptr, Hcb, FCHUNK,
           MROWS, FCHUNK, EMBD, GF_BIAS | GF_RELU | GF_BF16OUT);
      gemm(Hcb, FCHUNK, W2Tl + fc, FDIM, b2l, FB, nullptr, EMBD,
           MROWS, EMBD, FCHUNK, GF_F32OUT | ((fc == 0) ? GF_BIAS : GF_ACC));
    }
    // out = LN(ao + ff2)
    resln_kernel<<<MROWS / 8, 256, 0, stream>>>(AO, FB, ln3g + l * EMBD, ln3b + l * EMBD, OUT, OUTb);
    cur  = OUT;
    curb = OUTb;
  }

  // ----- decoder projection [4608,512] @ [512,127] (edge-guarded N) -----
  gemm(curb, EMBD, WdecT, EMBD, bdec, (float*)d_out, nullptr, EIN,
       MROWS, EIN, EMBD, GF_BIAS | GF_F32OUT);
  (void)cur;
}
